// UNet_1821066133518
// MI455X (gfx1250) — compile-verified, matrix operand order checked by reading, not run on hardware
//
#include <hip/hip_runtime.h>
#include <math.h>

// ---------------------------------------------------------------------------
// MI455X (gfx1250, wave32) implementation of the 3D attention U-Net.
// All GEMM-shaped work (3x3x3 convs via implicit GEMM, attention projections,
// k^T v outer products) runs on v_wmma_f32_16x16x32_f16 with f32 accumulate.
// f32 activations are converted to f16 while staging tiles into LDS.
// Staging is branchless (predicated offsets, unconditional loads) so the
// compiler can form load clauses instead of per-element exec-mask branches.
// ---------------------------------------------------------------------------

typedef __attribute__((ext_vector_type(16))) _Float16 v16h;
typedef __attribute__((ext_vector_type(8)))  _Float16 v8h;
typedef __attribute__((ext_vector_type(8)))  float    v8f;

// Input tensor order = flattened setup_inputs() dict order.
enum {
  IN_X = 0,
  IN_CONVIN_W, IN_CONVIN_B, IN_C11_W, IN_C11_B, IN_C12_W, IN_C12_B,
  IN_C2_W, IN_C2_B, IN_C3_W, IN_C3_B, IN_C4_W, IN_C4_B, IN_C5_W, IN_C5_B,
  IN_C6_W, IN_C6_B, IN_C7_W, IN_C7_B,
  IN_CA_WQ, IN_CA_WK, IN_CA_WV, IN_CA_WP, IN_CA_BP, IN_CA_SCALE, IN_CA_POS,
  IN_CA_DWCW, IN_CA_DWCB,
  IN_SA_WQ, IN_SA_WK, IN_SA_WV, IN_SA_WP, IN_SA_BP, IN_SA_SCALE, IN_SA_POS,
  IN_SA_DWCW, IN_SA_DWCB,
  IN_C6D_W, IN_C6D_B, IN_UP6_W, IN_UP6_B,
  IN_C5D_W, IN_C5D_B, IN_C4D_W, IN_C4D_B,
  IN_UP4_W, IN_UP4_B, IN_C3D_W, IN_C3D_B,
  IN_C2D_W, IN_C2D_B, IN_UP2_W, IN_UP2_B,
  IN_C1D_W, IN_C1D_B, IN_OUT_W
};

// Build a v16h WMMA fragment from two contiguous 8-half (16B) LDS reads.
// 16-bit fragment layout: lanes 0-15 hold K [hi*8..+7] and [16+hi*8..+7].
__device__ inline v16h ld_frag(const _Float16* p0, const _Float16* p1) {
  v8h lo = *(const v8h*)p0;
  v8h hi = *(const v8h*)p1;
  v16h r;
#pragma unroll
  for (int i = 0; i < 8; ++i) { r[i] = lo[i]; r[8 + i] = hi[i]; }
  return r;
}

// ---------------------------------------------------------------------------
// Implicit-GEMM 3x3x3 conv.  M = out channels, N = voxels, K = Cin*32
// (each input channel's 27 taps padded to a full 32-wide WMMA K-chunk, so
// tap arithmetic is compile-time after unrolling and the per-thread spatial
// decomposition is loop-invariant).  Dual input pointers fuse concat skips.
// Block = 128 thr = 4 waves; each wave owns two 16x16 tiles (A reuse),
// LDS double-buffered -> one barrier per K chunk, 2 WMMAs per barrier.
// Requires DHW % 128 == 0 (true for every level of this net).
// ---------------------------------------------------------------------------
__global__ __launch_bounds__(128) void conv3d_igemm_wmma(
    const float* __restrict__ in0, int C0, const float* __restrict__ in1, int C1,
    const float* __restrict__ wgt, const float* __restrict__ bias,
    float* __restrict__ out, int O, int D, int H, int W, int dil, int relu)
{
  const int DHW  = D * H * W;
  const int Cin  = C0 + C1;
  const int Kw   = Cin * 27;                 // true weight K extent
  const int n_base = blockIdx.x * 128;
  const int o_base = blockIdx.y * 16;
  const int tid  = threadIdx.x;
  const int lane = tid & 31, wid = tid >> 5;
  const int row  = lane & 15, hi = lane >> 4;

  __shared__ __align__(16) _Float16 ldsA[2][16 * 32];
  __shared__ __align__(16) _Float16 ldsB[2][128 * 40];

  // Loop-invariant spatial decomposition of this thread's im2col column.
  const int ng  = n_base + tid;
  const int ngc = (ng < DHW) ? ng : (DHW - 1);
  const int dsp = ngc / (H * W);
  const int hsp = (ngc / W) % H;
  const int wsp = ngc % W;

  // A-staging constants: k fixed per thread, o advances by 4 per step.
  const int kA  = tid & 31;
  const int oA  = tid >> 5;
  const bool kok = (kA < 27);

  v8f acc0 = {0, 0, 0, 0, 0, 0, 0, 0};
  v8f acc1 = {0, 0, 0, 0, 0, 0, 0, 0};

  for (int kb = 0; kb < Cin; ++kb) {       // one K-chunk per input channel
    const int buf = kb & 1;
    const float* __restrict__ src =
        (kb < C0) ? (in0 + (size_t)kb * DHW) : (in1 + (size_t)(kb - C0) * DHW);

    // --- stage A: 16 x 32 weight tile (taps 27..31 zero), branchless ---
#pragma unroll
    for (int t = 0; t < 4; ++t) {
      const int o  = oA + 4 * t;
      const bool ok = kok && ((o_base + o) < O);
      const size_t ofs = ok ? ((size_t)(o_base + o) * Kw + (size_t)kb * 27 + kA) : 0;
      float v = wgt[ofs];
      ldsA[buf][o * 32 + kA] = (_Float16)(ok ? v : 0.f);
    }

    // --- stage B: this thread's column, 27 real taps, branchless loads ---
#pragma unroll
    for (int t = 0; t < 32; ++t) {
      float v = 0.f;
      if (t < 27) {                          // compile-time after unroll
        const int kd = t / 9, kh = (t / 3) % 3, kw = t % 3;
        const int id = dsp + dil * (kd - 1);
        const int ih = hsp + dil * (kh - 1);
        const int iw = wsp + dil * (kw - 1);
        const bool ok = ((unsigned)id < (unsigned)D) &
                        ((unsigned)ih < (unsigned)H) &
                        ((unsigned)iw < (unsigned)W);
        const size_t ofs = ok ? (size_t)((id * H + ih) * W + iw) : 0;
        v = src[ofs];
        v = ok ? v : 0.f;
      }
      ldsB[buf][tid * 40 + t] = (_Float16)v;
    }

    // Prefetch next channel's neighborhood while this chunk is consumed.
    if (kb + 1 < Cin) {
      const float* nsrc = (kb + 1 < C0) ? (in0 + (size_t)(kb + 1) * DHW)
                                        : (in1 + (size_t)(kb + 1 - C0) * DHW);
      __builtin_prefetch(nsrc + ((size_t)dsp * H + hsp) * W + wsp, 0, 1);
    }

    __syncthreads();                          // single barrier per chunk
    v16h a  = ld_frag(&ldsA[buf][row * 32 + hi * 8],
                      &ldsA[buf][row * 32 + 16 + hi * 8]);
    const int nl0 = wid * 32 + row, nl1 = nl0 + 16;
    v16h b0 = ld_frag(&ldsB[buf][nl0 * 40 + hi * 8],
                      &ldsB[buf][nl0 * 40 + 16 + hi * 8]);
    v16h b1 = ld_frag(&ldsB[buf][nl1 * 40 + hi * 8],
                      &ldsB[buf][nl1 * 40 + 16 + hi * 8]);
    acc0 = __builtin_amdgcn_wmma_f32_16x16x32_f16(false, a, false, b0,
                                                  (short)0, acc0, false, false);
    acc1 = __builtin_amdgcn_wmma_f32_16x16x32_f16(false, a, false, b1,
                                                  (short)0, acc1, false, false);
  }

  const int n0 = n_base + wid * 32 + row;
  const int n1 = n0 + 16;
#pragma unroll
  for (int r = 0; r < 8; ++r) {
    const int o = o_base + r + 8 * hi;
    if (o < O) {
      const float bb = bias ? bias[o] : 0.f;
      if (n0 < DHW) {
        float v = acc0[r] + bb;
        if (relu) v = fmaxf(v, 0.f);
        out[(size_t)o * DHW + n0] = v;
      }
      if (n1 < DHW) {
        float v = acc1[r] + bb;
        if (relu) v = fmaxf(v, 0.f);
        out[(size_t)o * DHW + n1] = v;
      }
    }
  }
}

// ---------------------------------------------------------------------------
// Token projection GEMM:  out[n,j] = sum_k T[n,k]*W[j,k] (+addV[j]) (+addM[n,j])
// T: N x K row-major, W: J x K row-major, out: N x J row-major.
// Same wave tiling / double buffering as the conv kernel.
// ---------------------------------------------------------------------------
__global__ __launch_bounds__(128) void gemm_tok_wmma(
    const float* __restrict__ Tm, const float* __restrict__ Wm,
    const float* __restrict__ addV, const float* __restrict__ addM,
    float* __restrict__ out, int N, int J, int K)
{
  const int n_base = blockIdx.x * 128;
  const int j_base = blockIdx.y * 16;
  const int tid  = threadIdx.x;
  const int lane = tid & 31, wid = tid >> 5;
  const int row  = lane & 15, hi = lane >> 4;

  __shared__ __align__(16) _Float16 ldsA[2][16 * 32];
  __shared__ __align__(16) _Float16 ldsB[2][128 * 40];

  const int kS = tid & 31;        // staging k (fixed per thread)
  const int rS = tid >> 5;        // staging row base (+4 per step)

  v8f acc0 = {0, 0, 0, 0, 0, 0, 0, 0};
  v8f acc1 = {0, 0, 0, 0, 0, 0, 0, 0};

  const int nK = (K + 31) >> 5;
  for (int kb = 0; kb < nK; ++kb) {
    const int buf = kb & 1;
    const int kg  = (kb << 5) + kS;
    const bool kok = (kg < K);
#pragma unroll
    for (int t = 0; t < 4; ++t) {
      const int o = rS + 4 * t;
      const int j = j_base + o;
      const bool ok = kok && (j < J);
      const size_t ofs = ok ? ((size_t)j * K + kg) : 0;
      float v = Wm[ofs];
      ldsA[buf][o * 32 + kS] = (_Float16)(ok ? v : 0.f);
    }
#pragma unroll
    for (int t = 0; t < 32; ++t) {
      const int n  = rS + 4 * t;
      const int ngl = n_base + n;
      const bool ok = kok && (ngl < N);
      const size_t ofs = ok ? ((size_t)ngl * K + kg) : 0;
      float v = Tm[ofs];
      ldsB[buf][n * 40 + kS] = (_Float16)(ok ? v : 0.f);
    }
    __syncthreads();
    v16h a  = ld_frag(&ldsA[buf][row * 32 + hi * 8],
                      &ldsA[buf][row * 32 + 16 + hi * 8]);
    const int nl0 = wid * 32 + row, nl1 = nl0 + 16;
    v16h b0 = ld_frag(&ldsB[buf][nl0 * 40 + hi * 8],
                      &ldsB[buf][nl0 * 40 + 16 + hi * 8]);
    v16h b1 = ld_frag(&ldsB[buf][nl1 * 40 + hi * 8],
                      &ldsB[buf][nl1 * 40 + 16 + hi * 8]);
    acc0 = __builtin_amdgcn_wmma_f32_16x16x32_f16(false, a, false, b0,
                                                  (short)0, acc0, false, false);
    acc1 = __builtin_amdgcn_wmma_f32_16x16x32_f16(false, a, false, b1,
                                                  (short)0, acc1, false, false);
  }

  const int n0 = n_base + wid * 32 + row;
  const int n1 = n0 + 16;
#pragma unroll
  for (int r = 0; r < 8; ++r) {
    const int j = j_base + r + 8 * hi;
    if (j < J) {
      const float av = addV ? addV[j] : 0.f;
      if (n0 < N) {
        float v = acc0[r] + av;
        if (addM) v += addM[(size_t)n0 * J + j];
        out[(size_t)n0 * J + j] = v;
      }
      if (n1 < N) {
        float v = acc1[r] + av;
        if (addM) v += addM[(size_t)n1 * J + j];
        out[(size_t)n1 * J + j] = v;
      }
    }
  }
}

// ---------------------------------------------------------------------------
// kv[h,c,d] = sum_t k[t, h*hd+c] * v[t, h*hd+d]   (giant-K GEMM, hd x hd out)
// One wave per 16x16 tile, K split across gridDim.y, f32 atomicAdd reduce.
// Requires N % 32 == 0 (token counts 9216 / 131072).
// ---------------------------------------------------------------------------
__global__ __launch_bounds__(32) void kv_gemm_wmma(
    const float* __restrict__ km, const float* __restrict__ vm,
    float* __restrict__ kv, int N, int dim, int heads, int hd)
{
  const int MT = (hd + 15) >> 4;
  const int bx = blockIdx.x;
  const int h  = bx / (MT * MT);
  const int r  = bx % (MT * MT);
  const int mi = r / MT, ni = r % MT;
  const int lane = threadIdx.x;
  const int row = lane & 15, hi = lane >> 4;
  const int c = mi * 16 + row;   // A-matrix row (M)
  const int d = ni * 16 + row;   // B-matrix col (N)
  const bool cok = (c < hd), dok = (d < hd);
  const float* kb2 = km + h * hd;
  const float* vb2 = vm + h * hd;

  v8f acc = {0, 0, 0, 0, 0, 0, 0, 0};
  const int nchunks = N >> 5;
  for (int ch = blockIdx.y; ch < nchunks; ch += gridDim.y) {
    const int t0 = ch << 5;
    v16h a, b;
#pragma unroll
    for (int g = 0; g < 2; ++g)
#pragma unroll
      for (int j = 0; j < 8; ++j) {
        const int t = t0 + g * 16 + hi * 8 + j;
        const size_t aofs = cok ? ((size_t)t * dim + c) : 0;
        const size_t bofs = dok ? ((size_t)t * dim + d) : 0;
        float av = kb2[aofs]; av = cok ? av : 0.f;
        float bv = vb2[bofs]; bv = dok ? bv : 0.f;
        a[g * 8 + j] = (_Float16)av;
        b[g * 8 + j] = (_Float16)bv;
      }
    acc = __builtin_amdgcn_wmma_f32_16x16x32_f16(false, a, false, b,
                                                 (short)0, acc, false, false);
  }
#pragma unroll
  for (int rr = 0; rr < 8; ++rr) {
    const int m = mi * 16 + rr + 8 * hi;
    const int n = ni * 16 + (lane & 15);
    if (m < hd && n < hd)
      atomicAdd(&kv[((size_t)h * hd + m) * hd + n], acc[rr]);
  }
}

// ---------------------------------------------------------------------------
// Focused-attention row transform:  x = (relu(x)+1e-6)/softplus(scale);
// qn = ||x||; x = x^3; x = x/||x^3|| * qn.   One wave per token row.
// ---------------------------------------------------------------------------
__global__ __launch_bounds__(128) void qk_focus(
    float* __restrict__ q, const float* __restrict__ scale, int N, int dim)
{
  const int lane = threadIdx.x & 31;
  const int row  = blockIdx.x * 4 + (threadIdx.x >> 5);
  if (row >= N) return;
  float x[4] = {0.f, 0.f, 0.f, 0.f};
  const int cnt = (dim + 31) >> 5;
  float s2 = 0.f;
  for (int i = 0; i < cnt; ++i) {
    const int c = lane + (i << 5);
    if (c < dim) {
      float v = q[(size_t)row * dim + c];
      v = fmaxf(v, 0.f) + 1e-6f;
      const float s  = scale[c];
      const float sp = (s > 20.f) ? s : log1pf(expf(s));
      v /= sp;
      x[i] = v;
      s2 += v * v;
    }
  }
#pragma unroll
  for (int m = 16; m >= 1; m >>= 1) s2 += __shfl_xor(s2, m, 32);
  const float qn = sqrtf(s2);
  float s3 = 0.f;
  for (int i = 0; i < cnt; ++i) { x[i] = x[i] * x[i] * x[i]; s3 += x[i] * x[i]; }
#pragma unroll
  for (int m = 16; m >= 1; m >>= 1) s3 += __shfl_xor(s3, m, 32);
  const float sc = qn / sqrtf(fmaxf(s3, 1e-38f));
  for (int i = 0; i < cnt; ++i) {
    const int c = lane + (i << 5);
    if (c < dim) q[(size_t)row * dim + c] = x[i] * sc;
  }
}

// ksum[j] = sum_n k[n,j] ; one block per column.
__global__ __launch_bounds__(256) void ksum_kernel(
    const float* __restrict__ k, float* __restrict__ ksum, int N, int dim)
{
  const int j = blockIdx.x;
  float s = 0.f;
  for (int n = threadIdx.x; n < N; n += 256) s += k[(size_t)n * dim + j];
  __shared__ float red[256];
  red[threadIdx.x] = s;
  __syncthreads();
  for (int m = 128; m > 0; m >>= 1) {
    if ((int)threadIdx.x < m) red[threadIdx.x] += red[threadIdx.x + m];
    __syncthreads();
  }
  if (threadIdx.x == 0) ksum[j] = red[0];
}

// z[n,h] = 1 / (q[n, h*hd:].ksum[h*hd:] + 1e-6)
__global__ void z_kernel(const float* __restrict__ q, const float* __restrict__ ksum,
                         float* __restrict__ z, int N, int dim, int heads, int hd)
{
  const size_t idx = (size_t)blockIdx.x * blockDim.x + threadIdx.x;
  if (idx >= (size_t)N * heads) return;
  const int n = (int)(idx / heads), h = (int)(idx % heads);
  float s = 0.f;
  for (int c = 0; c < hd; ++c)
    s += q[(size_t)n * dim + h * hd + c] * ksum[h * hd + c];
  z[idx] = 1.f / (s + 1e-6f);
}

// out[n, h*hd+d] = (sum_c q[n,h*hd+c] * kv[h,c,d]) * z[n,h] + fm[n, h*hd+d]
__global__ void attn_out_kernel(const float* __restrict__ q, const float* __restrict__ kv,
                                const float* __restrict__ z, const float* __restrict__ fm,
                                float* __restrict__ out, int N, int dim, int heads, int hd)
{
  const size_t idx = (size_t)blockIdx.x * blockDim.x + threadIdx.x;
  if (idx >= (size_t)N * dim) return;
  const int n = (int)(idx / dim), j = (int)(idx % dim);
  const int h = j / hd, d = j % hd;
  float s = 0.f;
  for (int c = 0; c < hd; ++c)
    s += q[(size_t)n * dim + h * hd + c] * kv[((size_t)h * hd + c) * hd + d];
  out[idx] = s * z[(size_t)n * heads + h] + fm[idx];
}

// Depthwise 5x5x5 conv over v in token layout; fm written in token layout.
__global__ void dwconv5(const float* __restrict__ v, const float* __restrict__ w,
                        const float* __restrict__ b, float* __restrict__ fm,
                        int heads, int hd, int D0, int NU, int dim, size_t total)
{
  size_t idx = (size_t)blockIdx.x * blockDim.x + threadIdx.x;
  if (idx >= total) return;
  const int x = (int)(idx % NU); size_t t = idx / NU;
  const int y = (int)(t % NU); t /= NU;
  const int d0 = (int)(t % D0); t /= D0;
  const int c = (int)(t % hd);
  const int h = (int)(t / hd);
  float acc = b[c];
  for (int kd = 0; kd < 5; ++kd) {
    const int id = d0 + kd - 2;
    if ((unsigned)id >= (unsigned)D0) continue;
    for (int kh = 0; kh < 5; ++kh) {
      const int iy = y + kh - 2;
      if ((unsigned)iy >= (unsigned)NU) continue;
      for (int kw = 0; kw < 5; ++kw) {
        const int ix = x + kw - 2;
        if ((unsigned)ix >= (unsigned)NU) continue;
        const int n = (id * NU + iy) * NU + ix;
        acc += v[(size_t)n * dim + h * hd + c] * w[((c * 5 + kd) * 5 + kh) * 5 + kw];
      }
    }
  }
  const int n = (d0 * NU + y) * NU + x;
  fm[(size_t)n * dim + h * hd + c] = acc;
}

// Token pack/unpack between NCDHW feature map and (N_tokens, dim) layout.
// X layout: [C][S][NU][NU].  spectral=0: token n=(s*NU+y)*NU+x, feature c.
// spectral=1: token n=(c*NU+y)*NU+x, feature s.  dir=0: T=X ; dir=1: X*=T.
__global__ void tok_xpose(float* __restrict__ X, float* __restrict__ T,
                          int C, int S, int NU, int spectral, int dir)
{
  const size_t total = (size_t)C * S * NU * NU;
  size_t idx = (size_t)blockIdx.x * blockDim.x + threadIdx.x;
  if (idx >= total) return;
  const int x = (int)(idx % NU); size_t t = idx / NU;
  const int y = (int)(t % NU); t /= NU;
  const int s = (int)(t % S);
  const int c = (int)(t / S);
  size_t tok;
  if (spectral) tok = (size_t)((c * NU + y) * NU + x) * S + s;
  else          tok = (size_t)((s * NU + y) * NU + x) * C + c;
  if (dir == 0) T[tok] = X[idx];
  else          X[idx] *= T[tok];
}

__global__ void maxpool_hw_k(const float* __restrict__ in, float* __restrict__ out,
                             int C, int D, int H, int W)
{
  const int Ho = H >> 1, Wo = W >> 1;
  const size_t total = (size_t)C * D * Ho * Wo;
  size_t idx = (size_t)blockIdx.x * blockDim.x + threadIdx.x;
  if (idx >= total) return;
  const int x = (int)(idx % Wo); size_t t = idx / Wo;
  const int y = (int)(t % Ho); t /= Ho;
  const int d = (int)(t % D);
  const int c = (int)(t / D);
  const size_t base = (((size_t)c * D + d) * H + 2 * y) * W + 2 * x;
  out[idx] = fmaxf(fmaxf(in[base], in[base + 1]),
                   fmaxf(in[base + W], in[base + W + 1]));
}

// Transposed 1x2x2 conv, stride 2 in H,W, fused ReLU.
__global__ void deconv2x2(const float* __restrict__ x, const float* __restrict__ w,
                          const float* __restrict__ b, float* __restrict__ out,
                          int CI, int CO, int D, int H, int W)
{
  const size_t total = (size_t)CO * D * H * W;
  size_t idx = (size_t)blockIdx.x * blockDim.x + threadIdx.x;
  if (idx >= total) return;
  const int ww = (int)(idx % W); size_t t = idx / W;
  const int hh = (int)(t % H); t /= H;
  const int d = (int)(t % D);
  const int o = (int)(t / D);
  float a00 = 0.f, a01 = 0.f, a10 = 0.f, a11 = 0.f;
  for (int c = 0; c < CI; ++c) {
    const float xv = x[(((size_t)c * D + d) * H + hh) * W + ww];
    const float* wp = w + ((size_t)c * CO + o) * 4;
    a00 += xv * wp[0]; a01 += xv * wp[1];
    a10 += xv * wp[2]; a11 += xv * wp[3];
  }
  const float bb = b[o];
  const size_t ob = (((size_t)o * D + d) * (2 * H) + 2 * hh) * (2 * W) + 2 * ww;
  out[ob]             = fmaxf(a00 + bb, 0.f);
  out[ob + 1]         = fmaxf(a01 + bb, 0.f);
  out[ob + 2 * W]     = fmaxf(a10 + bb, 0.f);
  out[ob + 2 * W + 1] = fmaxf(a11 + bb, 0.f);
}

__global__ void add_residual(const float* __restrict__ a, const float* __restrict__ b,
                             float* __restrict__ o, size_t n)
{
  size_t idx = (size_t)blockIdx.x * blockDim.x + threadIdx.x;
  if (idx < n) o[idx] = a[idx] + b[idx];
}

// ---------------------------------------------------------------------------
static inline unsigned cdivz(size_t a, size_t b) { return (unsigned)((a + b - 1) / b); }

static void run_focused_attn(
    const float* T, int N, int dim, int heads, int D0, int NU,
    const float* wq, const float* wk, const float* wv, const float* wp,
    const float* bp, const float* scale, const float* pos,
    const float* dwcw, const float* dwcb,
    float* q, float* k, float* v, float* fm, float* aout,
    float* ksum, float* z, float* kvbuf, float* result, hipStream_t stream)
{
  const int hd = dim / heads;
  dim3 gb(cdivz(N, 128), cdivz(dim, 16));
  gemm_tok_wmma<<<gb, 128, 0, stream>>>(T, wq, nullptr, nullptr, q, N, dim, dim);
  gemm_tok_wmma<<<gb, 128, 0, stream>>>(T, wk, nullptr, pos,     k, N, dim, dim);
  gemm_tok_wmma<<<gb, 128, 0, stream>>>(T, wv, nullptr, nullptr, v, N, dim, dim);
  qk_focus<<<cdivz(N, 4), 128, 0, stream>>>(q, scale, N, dim);
  qk_focus<<<cdivz(N, 4), 128, 0, stream>>>(k, scale, N, dim);
  ksum_kernel<<<dim, 256, 0, stream>>>(k, ksum, N, dim);
  z_kernel<<<cdivz((size_t)N * heads, 256), 256, 0, stream>>>(q, ksum, z, N, dim, heads, hd);
  hipMemsetAsync(kvbuf, 0, sizeof(float) * (size_t)heads * hd * hd, stream);
  const int MT = (hd + 15) / 16;
  dim3 gkv((unsigned)(heads * MT * MT), 64);
  kv_gemm_wmma<<<gkv, 32, 0, stream>>>(k, v, kvbuf, N, dim, heads, hd);
  const size_t fmtot = (size_t)heads * hd * D0 * NU * NU;   // == N*dim
  dwconv5<<<cdivz(fmtot, 256), 256, 0, stream>>>(v, dwcw, dwcb, fm, heads, hd, D0, NU, dim, fmtot);
  attn_out_kernel<<<cdivz((size_t)N * dim, 256), 256, 0, stream>>>(q, kvbuf, z, fm, aout, N, dim, heads, hd);
  gemm_tok_wmma<<<gb, 128, 0, stream>>>(aout, wp, bp, nullptr, result, N, dim, dim);
}

// ---------------------------------------------------------------------------
extern "C" void kernel_launch(void* const* d_in, const int* in_sizes, int n_in,
                              void* d_out, int out_size, void* d_ws, size_t ws_size,
                              hipStream_t stream)
{
  (void)in_sizes; (void)n_in; (void)out_size; (void)ws_size;
  auto F = [&](int i) { return (const float*)d_in[i]; };

  float* ws = (float*)d_ws;
  size_t off = 0;
  auto alloc = [&](size_t n) { float* p = ws + off; off += n; return p; };

  const int Dd = 9;
  const size_t S256 = (size_t)Dd * 256 * 256;  // 589824
  const size_t S128 = (size_t)Dd * 128 * 128;
  const size_t S64  = (size_t)Dd * 64 * 64;
  const size_t S32  = (size_t)Dd * 32 * 32;    // 9216

  // Bump allocation from workspace (peak ~358 MB of f32; L2-resident at the
  // hot decoder levels).  A few dead buffers are recycled explicitly.
  float* hxin = alloc(16 * S256);
  float* t1   = alloc(16 * S256);
  float* hx1  = alloc(16 * S256);
  float* p1   = alloc(16 * S128);
  float* hx2  = alloc(32 * S128);
  float* hx3  = alloc(32 * S128);
  float* p2   = alloc(32 * S64);
  float* hx4  = alloc(64 * S64);
  float* hx5  = alloc(64 * S64);
  float* p3   = alloc(64 * S32);
  float* hx6  = alloc(128 * S32);
  float* hx7  = alloc(128 * S32);
  const size_t TD = 1179648;         // max tokens*dim (both attentions)
  float* tok  = alloc(TD);
  float* qb   = alloc(TD);
  float* kb   = alloc(TD);
  float* vb   = alloc(TD);
  float* fmb  = alloc(TD);
  float* aout = alloc(TD);
  float* resb = alloc(TD);
  float* zb   = alloc((size_t)131072 * 3);
  float* ksum = alloc(128);
  float* kvb  = alloc(2 * 64 * 64);
  float* hx6d   = alloc(128 * S32);
  float* hx6dup = alloc(64 * S64);
  float* hx5d   = alloc(64 * S64);
  float* hx4d   = alloc(64 * S64);
  float* hx4dup = alloc(32 * S128);
  float* hx3d   = alloc(32 * S128);
  float* hx2d   = alloc(32 * S128);
  float* hx2dup = t1;                 // c1 temp is dead by now
  float* hx1d   = alloc(16 * S256);
  float* outc   = p3;                 // p3 dead after hx6

  auto conv = [&](const float* i0, int C0, const float* i1, int C1,
                  int wi, int bi, float* o, int O, int HW, int dil, int relu) {
    dim3 g(cdivz((size_t)Dd * HW * HW, 128), cdivz(O, 16));
    conv3d_igemm_wmma<<<g, 128, 0, stream>>>(
        i0, C0, i1, C1, F(wi), bi >= 0 ? F(bi) : nullptr, o, O, Dd, HW, HW, dil, relu);
  };

  const float* x = F(IN_X);

  // ---------------- encoder ----------------
  conv(x,    1,  nullptr, 0, IN_CONVIN_W, IN_CONVIN_B, hxin, 16, 256, 1, 1);
  conv(hxin, 16, nullptr, 0, IN_C11_W,    IN_C11_B,    t1,   16, 256, 1, 1);
  conv(t1,   16, nullptr, 0, IN_C12_W,    IN_C12_B,    hx1,  16, 256, 1, 1);
  maxpool_hw_k<<<cdivz(16 * S128, 256), 256, 0, stream>>>(hx1, p1, 16, Dd, 256, 256);
  conv(p1,   16, nullptr, 0, IN_C2_W, IN_C2_B, hx2, 32, 128, 1, 1);
  conv(hx2,  32, nullptr, 0, IN_C3_W, IN_C3_B, hx3, 32, 128, 1, 1);
  maxpool_hw_k<<<cdivz(32 * S64, 256), 256, 0, stream>>>(hx3, p2, 32, Dd, 128, 128);
  conv(p2,   32, nullptr, 0, IN_C4_W, IN_C4_B, hx4, 64, 64, 1, 1);
  conv(hx4,  64, nullptr, 0, IN_C5_W, IN_C5_B, hx5, 64, 64, 1, 1);
  maxpool_hw_k<<<cdivz(64 * S32, 256), 256, 0, stream>>>(hx5, p3, 64, Dd, 64, 64);
  conv(p3,   64, nullptr, 0, IN_C6_W, IN_C6_B, hx6, 128, 32, 1, 1);
  conv(hx6, 128, nullptr, 0, IN_C7_W, IN_C7_B, hx7, 128, 32, 2, 1);  // dilated

  // ---------------- channel attention (dim=128, heads=2) ----------------
  const unsigned TXG = cdivz(128 * S32, 256);
  tok_xpose<<<TXG, 256, 0, stream>>>(hx7, tok, 128, 9, 32, 0, 0);
  run_focused_attn(tok, 9216, 128, 2, 9, 32,
                   F(IN_CA_WQ), F(IN_CA_WK), F(IN_CA_WV), F(IN_CA_WP),
                   F(IN_CA_BP), F(IN_CA_SCALE), F(IN_CA_POS),
                   F(IN_CA_DWCW), F(IN_CA_DWCB),
                   qb, kb, vb, fmb, aout, ksum, zb, kvb, resb, stream);
  tok_xpose<<<TXG, 256, 0, stream>>>(hx7, resb, 128, 9, 32, 0, 1);

  // ---------------- spectral attention (dim=9, heads=3) ----------------
  tok_xpose<<<TXG, 256, 0, stream>>>(hx7, tok, 128, 9, 32, 1, 0);
  run_focused_attn(tok, 131072, 9, 3, 128, 32,
                   F(IN_SA_WQ), F(IN_SA_WK), F(IN_SA_WV), F(IN_SA_WP),
                   F(IN_SA_BP), F(IN_SA_SCALE), F(IN_SA_POS),
                   F(IN_SA_DWCW), F(IN_SA_DWCB),
                   qb, kb, vb, fmb, aout, ksum, zb, kvb, resb, stream);
  tok_xpose<<<TXG, 256, 0, stream>>>(hx7, resb, 128, 9, 32, 1, 1);

  // ---------------- decoder (concats fused via dual-input conv) ----------------
  conv(hx7, 128, hx6, 128, IN_C6D_W, IN_C6D_B, hx6d, 128, 32, 1, 1);
  deconv2x2<<<cdivz(64 * S32, 256), 256, 0, stream>>>(
      hx6d, F(IN_UP6_W), F(IN_UP6_B), hx6dup, 128, 64, Dd, 32, 32);
  conv(hx6dup, 64, hx5, 64, IN_C5D_W, IN_C5D_B, hx5d, 64, 64, 1, 1);
  conv(hx5d,   64, hx4, 64, IN_C4D_W, IN_C4D_B, hx4d, 64, 64, 1, 1);
  deconv2x2<<<cdivz(32 * S64, 256), 256, 0, stream>>>(
      hx4d, F(IN_UP4_W), F(IN_UP4_B), hx4dup, 64, 32, Dd, 64, 64);
  conv(hx4dup, 32, hx3, 32, IN_C3D_W, IN_C3D_B, hx3d, 32, 128, 1, 1);
  conv(hx3d,   32, hx2, 32, IN_C2D_W, IN_C2D_B, hx2d, 32, 128, 1, 1);
  deconv2x2<<<cdivz(16 * S128, 256), 256, 0, stream>>>(
      hx2d, F(IN_UP2_W), F(IN_UP2_B), hx2dup, 32, 16, Dd, 128, 128);
  conv(hx2dup, 16, hx1,  16, IN_C1D_W, IN_C1D_B, hx1d, 16, 256, 1, 1);
  conv(hx1d,   16, hxin, 16, IN_OUT_W, -1,       outc, 1,  256, 1, 0);

  add_residual<<<cdivz(S256, 256), 256, 0, stream>>>(outc, x, (float*)d_out, S256);
}
